// MambaModel_63685775065214
// MI455X (gfx1250) — matrix operand order repaired and checked by reading, not compiled, not run on hardware
//
#include <hip/hip_runtime.h>
#include <hip/hip_bf16.h>
#include <math.h>

#define BATCH  16
#define SEQL   8192
#define TOK    (BATCH*SEQL)     /* 131072 tokens */
#define DMODEL 128
#define DINNER 256
#define DSTATE 16
#define LN_EPS 1e-5f

typedef __bf16 bf16_t;
typedef __attribute__((ext_vector_type(16))) __bf16 bf16x16;
typedef __attribute__((ext_vector_type(8)))  __bf16 bf16x8;
typedef __attribute__((ext_vector_type(4)))  __bf16 bf16x4;
typedef __attribute__((ext_vector_type(8)))  float  f32x8;

__device__ __forceinline__ float fast_rcp(float x)  { return __builtin_amdgcn_rcpf(x); }
__device__ __forceinline__ float sigmoidf_(float x) { return fast_rcp(1.f + __expf(-x)); }
__device__ __forceinline__ float siluf(float x)     { return x * sigmoidf_(x); }
__device__ __forceinline__ float softplusf(float x) { return (x > 20.f) ? x : __logf(1.f + __expf(x)); }

union ABits { bf16x16 f; bf16x8 h[2]; };

// ---------------------------------------------------------------------------
// Core: one wave computes a 16*MT(M) x 16*NT(N) tile, K-loop of 32, bf16 WMMA.
//   A: row-major [M][K] activations (bf16), Arow pre-offset to row block.
//   B: row-major [K][N] weights (bf16), Bcol pre-offset to column block.
// A frag: lane l -> row (l&15); halves K = g*8..+7 and 16+g*8..+7 (g=l>>4)
// B frag: lane l -> column (l&15); halves K = g*8..+7 and 16+g*8..+7 (g=l>>4)
// Each B fragment is reused by MT WMMAs -> fewer load-waits per WMMA.
// ---------------------------------------------------------------------------
template<int MT, int NT>
__device__ __forceinline__ void wmma_tile(
    const bf16_t* __restrict__ Arow, int lda,
    const bf16_t* __restrict__ Bcol, int ldb, int K,
    f32x8 (&acc)[MT][NT])
{
  const int lane = threadIdx.x & 31;
  const int r = lane & 15;
  const int g = lane >> 4;
  const bf16_t* pa = Arow + r * lda + g * 8;
  const bf16_t* pb = Bcol + r + g * 8 * ldb;
  for (int k0 = 0; k0 < K; k0 += 32) {
    ABits a[MT];
#pragma unroll
    for (int mt = 0; mt < MT; ++mt) {
      a[mt].h[0] = *(const bf16x8*)(pa + mt*16*lda + k0);
      a[mt].h[1] = *(const bf16x8*)(pa + mt*16*lda + k0 + 16);
    }
    const bf16_t* pbk = pb + k0 * ldb;
#pragma unroll
    for (int nt = 0; nt < NT; ++nt) {
      ABits b;
#pragma unroll
      for (int s = 0; s < 8; ++s) {
        b.f[s]     = pbk[nt*16 + s*ldb];
        b.f[8 + s] = pbk[nt*16 + (16 + s)*ldb];
      }
#pragma unroll
      for (int mt = 0; mt < MT; ++mt)
        acc[mt][nt] = __builtin_amdgcn_wmma_f32_16x16x32_bf16(false, a[mt].f, false, b.f,
                                                              (short)0, acc[mt][nt],
                                                              false, false);
    }
  }
}

template<int MT, int NT>
__device__ __forceinline__ void zero_acc(f32x8 (&acc)[MT][NT])
{
#pragma unroll
  for (int mt = 0; mt < MT; ++mt)
#pragma unroll
    for (int nt = 0; nt < NT; ++nt)
      acc[mt][nt] = (f32x8){0,0,0,0,0,0,0,0};
}

// ---------------------------------------------------------------------------
// GEMM1: xz = x @ in_proj_w + b ; x_p -> silu -> U (f32) + XC (bf16); z -> silu -> ZS
// Wave tile 32x64 (MT=2, NT=4), colTiles = 512/64 = 8, rowTiles = TOK/32.
// ---------------------------------------------------------------------------
__global__ void gemm1_inproj(const bf16_t* __restrict__ A,   /* [TOK][128] */
                             const bf16_t* __restrict__ W,   /* [128][512] */
                             const float*  __restrict__ bias,/* [512] */
                             float*  __restrict__ U,         /* [TOK][256] silu(x_p) */
                             bf16_t* __restrict__ XC,        /* [TOK][256] bf16 copy */
                             float*  __restrict__ ZS)        /* [TOK][256] silu(z) */
{
  const int colTiles = 512/64;
  const int w = blockIdx.x * (blockDim.x >> 5) + (threadIdx.x >> 5);
  if (w >= (TOK/32)*colTiles) return;
  const int rowT = w / colTiles, colT = w % colTiles;
  f32x8 acc[2][4];
  zero_acc(acc);
  wmma_tile<2,4>(A + rowT*32*DMODEL, DMODEL, W + colT*64, 512, DMODEL, acc);
  const int lane = threadIdx.x & 31;
  const int n0   = colT*64 + (lane & 15);
#pragma unroll
  for (int mt = 0; mt < 2; ++mt) {
    const int row0 = rowT*32 + mt*16 + (lane >> 4) * 8;
#pragma unroll
    for (int r = 0; r < 8; ++r) {
      const int row = row0 + r;
#pragma unroll
      for (int nt = 0; nt < 4; ++nt) {
        const int j = n0 + 16*nt;
        const float v = acc[mt][nt][r] + bias[j];
        if (j < DINNER) {                 // x_p half (wave-uniform: colT<4)
          const float s = siluf(v);
          U [row*DINNER + j] = s;
          XC[row*DINNER + j] = (bf16_t)s;
        } else {                          // z half
          ZS[row*DINNER + (j - DINNER)] = siluf(v);
        }
      }
    }
  }
}

// ---------------------------------------------------------------------------
// GEMM2: fused x_proj (N=32) + dt_proj (N=256) -> Bc, Cc, softplus(dt)
// Wave tile 32x32 (MT=2, NT=2), colTiles = 288/32 = 9.
// ---------------------------------------------------------------------------
__global__ void gemm2_xdt(const bf16_t* __restrict__ A,     /* [TOK][256] */
                          const bf16_t* __restrict__ W,     /* [256][288] */
                          const float*  __restrict__ xp_b,  /* [32] */
                          const float*  __restrict__ dt_b,  /* [256] */
                          float* __restrict__ Bc,           /* [TOK][16] */
                          float* __restrict__ Cc,           /* [TOK][16] */
                          float* __restrict__ DT)           /* [TOK][256] */
{
  const int colTiles = 288/32;  // 9
  const int w = blockIdx.x * (blockDim.x >> 5) + (threadIdx.x >> 5);
  if (w >= (TOK/32)*colTiles) return;
  const int rowT = w / colTiles, colT = w % colTiles;
  f32x8 acc[2][2];
  zero_acc(acc);
  wmma_tile<2,2>(A + rowT*32*DINNER, DINNER, W + colT*32, 288, DINNER, acc);
  const int lane = threadIdx.x & 31;
  const int n0   = colT*32 + (lane & 15);
#pragma unroll
  for (int mt = 0; mt < 2; ++mt) {
    const int row0 = rowT*32 + mt*16 + (lane >> 4) * 8;
#pragma unroll
    for (int r = 0; r < 8; ++r) {
      const int row = row0 + r;
#pragma unroll
      for (int nt = 0; nt < 2; ++nt) {
        const int j = n0 + 16*nt;
        const float v = acc[mt][nt][r];
        if (j < DSTATE) {                       // B (wave-uniform per nt)
          Bc[row*DSTATE + j] = v + xp_b[j];
        } else if (j < 2*DSTATE) {              // C
          Cc[row*DSTATE + (j - DSTATE)] = v + xp_b[j];
        } else {                                // dt -> softplus
          DT[row*DINNER + (j - 2*DSTATE)] = softplusf(v + dt_b[j - 2*DSTATE]);
        }
      }
    }
  }
}

// ---------------------------------------------------------------------------
// GEMM3: out = y @ out_proj_w + b -> TMP [TOK][128]
// Wave tile 32x64 (MT=2, NT=4), colTiles = 128/64 = 2.
// ---------------------------------------------------------------------------
__global__ void gemm3_outproj(const bf16_t* __restrict__ A, /* [TOK][256] */
                              const bf16_t* __restrict__ W, /* [256][128] */
                              const float*  __restrict__ bias,
                              float* __restrict__ OUT)      /* [TOK][128] */
{
  const int colTiles = 128/64;  // 2
  const int w = blockIdx.x * (blockDim.x >> 5) + (threadIdx.x >> 5);
  if (w >= (TOK/32)*colTiles) return;
  const int rowT = w / colTiles, colT = w % colTiles;
  f32x8 acc[2][4];
  zero_acc(acc);
  wmma_tile<2,4>(A + rowT*32*DINNER, DINNER, W + colT*64, DMODEL, DINNER, acc);
  const int lane = threadIdx.x & 31;
  const int n0   = colT*64 + (lane & 15);
#pragma unroll
  for (int mt = 0; mt < 2; ++mt) {
    const int row0 = rowT*32 + mt*16 + (lane >> 4) * 8;
#pragma unroll
    for (int r = 0; r < 8; ++r) {
      const int row = row0 + r;
#pragma unroll
      for (int nt = 0; nt < 4; ++nt) {
        const int j = n0 + 16*nt;
        OUT[row*DMODEL + j] = acc[mt][nt][r] + bias[j];
      }
    }
  }
}

// ---------------------------------------------------------------------------
// Selective scan: one lane per (channel, state). Wave = 2 channels x 16 states.
// y overwrites U in place. Cross-state sum via 4-step shfl_xor butterfly.
// ---------------------------------------------------------------------------
__global__ void scan_kernel(float* __restrict__ UY,        /* [TOK][256] in: u, out: y */
                            const float* __restrict__ DT,  /* [TOK][256] */
                            const float* __restrict__ Bc,  /* [TOK][16] */
                            const float* __restrict__ Cc,  /* [TOK][16] */
                            const float* __restrict__ A_log,/* [256][16] */
                            const float* __restrict__ Dv)  /* [256] */
{
  const int lane = threadIdx.x & 31;
  const int s = lane & 15;
  const int wv = blockIdx.x * (blockDim.x >> 5) + (threadIdx.x >> 5);
  if (wv >= BATCH * (DINNER/2)) return;
  const int b = wv / (DINNER/2);
  const int d = (wv % (DINNER/2)) * 2 + (lane >> 4);
  const float Ac = -__expf(A_log[d*DSTATE + s]);
  const float Dc = Dv[d];
  float h = 0.f;
  const long base = (long)b * SEQL;
  for (int t = 0; t < SEQL; ++t) {
    const long tok = base + t;
    const float u  = UY[tok*DINNER + d];
    const float dt = DT[tok*DINNER + d];
    const float Bs = Bc[tok*DSTATE + s];
    const float Cs = Cc[tok*DSTATE + s];
    const float dA = __expf(dt * Ac);
    h = fmaf(h, dA, dt * u * Bs);
    float p = h * Cs;
    p += __shfl_xor(p, 1, 32);
    p += __shfl_xor(p, 2, 32);
    p += __shfl_xor(p, 4, 32);
    p += __shfl_xor(p, 8, 32);
    if (s == 0) UY[tok*DINNER + d] = fmaf(Dc, u, p);
  }
}

// ---------------------------------------------------------------------------
// Elementwise helpers
// ---------------------------------------------------------------------------
__global__ void ymul_to_bf16(const float* __restrict__ Y, const float* __restrict__ ZS,
                             bf16_t* __restrict__ O, long n)
{
  const long i = (long)blockIdx.x * blockDim.x + threadIdx.x;
  if (i < n) O[i] = (bf16_t)(Y[i] * ZS[i]);
}

__global__ void cast_bf16_2d(const float* __restrict__ s, int sld,
                             bf16_t* __restrict__ d, int dld, int rows, int cols)
{
  const int i = blockIdx.x * blockDim.x + threadIdx.x;
  if (i >= rows*cols) return;
  const int r = i / cols, c = i - r*cols;
  d[r*dld + c] = (bf16_t)s[r*sld + c];
}

// Embedding + NaN-mask attention gate: X = (f@emb_w + eb) * sigmoid(m@mask_w + mb)
// Also emits bf16 copy for GEMM1.
__global__ void embed_kernel(const float* __restrict__ F,
                             const float* __restrict__ ew, const float* __restrict__ eb,
                             const float* __restrict__ mw, const float* __restrict__ mb,
                             float* __restrict__ X, bf16_t* __restrict__ XB)
{
  const long i = (long)blockIdx.x * blockDim.x + threadIdx.x;
  if (i >= (long)TOK * DMODEL) return;
  const int  col = (int)(i & (DMODEL - 1));
  const long tok = i >> 7;
  const float* f = F + tok * 8;
  float e = eb[col], m = mb[col];
#pragma unroll
  for (int k = 0; k < 8; ++k) {
    const float vv = f[k];
    const bool nn = (vv != vv);           // isnan
    e += (nn ? 0.f : vv)  * ew[k*DMODEL + col];
    m += (nn ? 0.f : 1.f) * mw[k*DMODEL + col];
  }
  const float x = e * sigmoidf_(m);
  X[i]  = x;
  XB[i] = (bf16_t)x;
}

// Residual add + LayerNorm over 128 features; wave-per-token, lane holds 4 feats.
// Also emits bf16 copy of the new activations for the next layer's GEMM1.
__global__ void resid_ln(const float* __restrict__ TMP, float* __restrict__ X,
                         bf16_t* __restrict__ XB,
                         const float* __restrict__ w, const float* __restrict__ bb)
{
  const int wv = blockIdx.x * (blockDim.x >> 5) + (threadIdx.x >> 5);
  if (wv >= TOK) return;
  const int lane = threadIdx.x & 31;
  const float4 t = *(const float4*)(TMP + (long)wv*DMODEL + lane*4);
  const float4 x = *(const float4*)(X   + (long)wv*DMODEL + lane*4);
  float v[4] = {t.x + x.x, t.y + x.y, t.z + x.z, t.w + x.w};
  float sum = v[0] + v[1] + v[2] + v[3];
#pragma unroll
  for (int m = 16; m >= 1; m >>= 1) sum += __shfl_xor(sum, m, 32);
  const float mu = sum * (1.f/128.f);
  float q = 0.f;
#pragma unroll
  for (int i = 0; i < 4; ++i) { v[i] -= mu; q += v[i]*v[i]; }
#pragma unroll
  for (int m = 16; m >= 1; m >>= 1) q += __shfl_xor(q, m, 32);
  const float inv = rsqrtf(q * (1.f/128.f) + LN_EPS);
  float o[4];
#pragma unroll
  for (int i = 0; i < 4; ++i) o[i] = v[i]*inv*w[lane*4+i] + bb[lane*4+i];
  *(float4*)(X + (long)wv*DMODEL + lane*4) = make_float4(o[0], o[1], o[2], o[3]);
  bf16x4 ob;
#pragma unroll
  for (int i = 0; i < 4; ++i) ob[i] = (bf16_t)o[i];
  *(bf16x4*)(XB + (long)wv*DMODEL + lane*4) = ob;   // 8-byte store
}

// Head on last token of each batch: relu(p@W1+b1) @ W2 + b2 -> tanh
__global__ void head_kernel(const float* __restrict__ X,
                            const float* __restrict__ w1, const float* __restrict__ b1,
                            const float* __restrict__ w2, const float* __restrict__ b2,
                            float* __restrict__ out)
{
  __shared__ float sh[64];
  const int b = blockIdx.x, j = threadIdx.x;
  const float* p = X + ((long)b*SEQL + (SEQL-1)) * DMODEL;
  float acc = b1[j];
  for (int i = 0; i < DMODEL; ++i) acc += p[i] * w1[i*64 + j];
  acc = fmaxf(acc, 0.f);
  sh[j] = acc * w2[j];
  __syncthreads();
  for (int s = 32; s > 0; s >>= 1) { if (j < s) sh[j] += sh[j + s]; __syncthreads(); }
  if (j == 0) out[b] = tanhf(sh[0] + b2[0]);
}

// ---------------------------------------------------------------------------
extern "C" void kernel_launch(void* const* d_in, const int* in_sizes, int n_in,
                              void* d_out, int out_size, void* d_ws, size_t ws_size,
                              hipStream_t stream)
{
  (void)in_sizes; (void)n_in; (void)out_size; (void)ws_size;
  const float* features  = (const float*)d_in[1];
  const float* emb_w     = (const float*)d_in[2];
  const float* emb_b     = (const float*)d_in[3];
  const float* mask_w    = (const float*)d_in[4];
  const float* mask_b    = (const float*)d_in[5];
  const float* in_proj_w = (const float*)d_in[6];
  const float* in_proj_b = (const float*)d_in[7];
  const float* x_proj_w  = (const float*)d_in[8];
  const float* x_proj_b  = (const float*)d_in[9];
  const float* dt_proj_w = (const float*)d_in[10];
  const float* dt_proj_b = (const float*)d_in[11];
  const float* out_proj_w= (const float*)d_in[12];
  const float* out_proj_b= (const float*)d_in[13];
  const float* A_log     = (const float*)d_in[14];
  const float* Dvec      = (const float*)d_in[15];
  const float* ln_w      = (const float*)d_in[16];
  const float* ln_b      = (const float*)d_in[17];
  const float* head1_w   = (const float*)d_in[18];
  const float* head1_b   = (const float*)d_in[19];
  const float* head2_w   = (const float*)d_in[20];
  const float* head2_b   = (const float*)d_in[21];

  // Workspace partition (256B aligned)
  char* p = (char*)d_ws;
  auto alloc = [&](size_t bytes) -> void* {
    void* q = (void*)p; p += (bytes + 255) & ~(size_t)255; return q;
  };
  float*  X   = (float*) alloc((size_t)TOK*DMODEL*4);  // activations / residual
  bf16_t* XB1 = (bf16_t*)alloc((size_t)TOK*DMODEL*2);  // x bf16 for GEMM1
  bf16_t* XB2 = (bf16_t*)alloc((size_t)TOK*DINNER*2);  // xc bf16 / y bf16
  float*  U   = (float*) alloc((size_t)TOK*DINNER*4);  // u = silu(x_p); overwritten by y
  float*  ZS  = (float*) alloc((size_t)TOK*DINNER*4);  // silu(z)
  float*  DTb = (float*) alloc((size_t)TOK*DINNER*4);  // softplus(dt); reused as GEMM3 tmp
  float*  Bc  = (float*) alloc((size_t)TOK*DSTATE*4);
  float*  Cc  = (float*) alloc((size_t)TOK*DSTATE*4);
  bf16_t* W1  = (bf16_t*)alloc(128*512*2);
  bf16_t* W2  = (bf16_t*)alloc(256*288*2);
  bf16_t* W3  = (bf16_t*)alloc(256*128*2);

  embed_kernel<<<(TOK*DMODEL)/256, 256, 0, stream>>>(features, emb_w, emb_b,
                                                     mask_w, mask_b, X, XB1);

  for (int l = 0; l < 2; ++l) {
    // Weight casts to bf16 (layout preserved: WMMA B-frag reads row-major [K][N])
    cast_bf16_2d<<<(128*512)/256, 256, 0, stream>>>(in_proj_w + l*128*512, 512, W1, 512, 128, 512);
    cast_bf16_2d<<<(256*32 +255)/256, 256, 0, stream>>>(x_proj_w  + l*256*32,  32,  W2,      288, 256, 32);
    cast_bf16_2d<<<(256*256)/256, 256, 0, stream>>>(dt_proj_w + l*256*256, 256, W2 + 32, 288, 256, 256);
    cast_bf16_2d<<<(256*128)/256, 256, 0, stream>>>(out_proj_w+ l*256*128, 128, W3, 128, 256, 128);

    gemm1_inproj<<<(TOK/32)*(512/64)/8, 256, 0, stream>>>(XB1, W1, in_proj_b + l*512,
                                                          U, XB2, ZS);
    gemm2_xdt<<<(TOK/32)*(288/32)/8, 256, 0, stream>>>(XB2, W2, x_proj_b + l*32,
                                                       dt_proj_b + l*256, Bc, Cc, DTb);
    scan_kernel<<<(BATCH*(DINNER/2))/8, 256, 0, stream>>>(U, DTb, Bc, Cc,
                                                          A_log + l*DINNER*DSTATE,
                                                          Dvec + l*DINNER);
    ymul_to_bf16<<<((long)TOK*DINNER)/256, 256, 0, stream>>>(U, ZS, XB2, (long)TOK*DINNER);
    gemm3_outproj<<<(TOK/32)*(128/64)/8, 256, 0, stream>>>(XB2, W3, out_proj_b + l*128, DTb);
    resid_ln<<<TOK/8, 256, 0, stream>>>(DTb, X, XB1, ln_w + l*DMODEL, ln_b + l*DMODEL);
  }

  head_kernel<<<BATCH, 64, 0, stream>>>(X, head1_w, head1_b, head2_w, head2_b,
                                        (float*)d_out);
}
